// LSTMAttentionBlock_13288628814041
// MI455X (gfx1250) — matrix operand order repaired and checked by reading, not compiled, not run on hardware
//
#include <hip/hip_runtime.h>
#include <hip/hip_bf16.h>

// ---------------------------------------------------------------------------
// LSTM + self-attention block for MI455X (gfx1250, wave32, WMMA).
// f16 WMMA (v_wmma_f32_16x16x32_f16) with fp32 accumulation everywhere.
// WMMA inner loops are software-pipelined one k-tile ahead; sched_barrier(0)
// pins the prefetch->compute order so the backend cannot rotate the pipeline
// back to load->wait(0)->wmma.
// ---------------------------------------------------------------------------

typedef _Float16 half_t;
typedef __attribute__((ext_vector_type(16))) _Float16 v16h;
typedef __attribute__((ext_vector_type(8)))  _Float16 v8h;
typedef __attribute__((ext_vector_type(8)))  float    v8f;

#define DEV __device__ __forceinline__

#define Bz    16
#define Tz    512
#define Dz    512
#define FH    2048   // 4 * UNITS
#define NHEAD 8
#define HD    8
#define QKVz  64

// Scheduling fence: keep prefetch loads (issued before) from sinking past the
// WMMAs (issued after). Emits no machine instruction.
DEV void pipeline_fence() { __builtin_amdgcn_sched_barrier(0); }

// ---- WMMA fragment loaders (CDNA5 ISA 7.12.2 layouts, wave32) --------------
// A-matrix 16x32 f16: lane L: row = L%16; halves K = (L/16)*8 + {0..7} and +16.
DEV v16h load_a_frag(const half_t* base, int ld) {
  int l = threadIdx.x & 31;
  const half_t* p = base + (l & 15) * ld + (l >> 4) * 8;
  v8h lo = *(const v8h*)p;
  v8h hi = *(const v8h*)(p + 16);
  v16h a;
#pragma unroll
  for (int i = 0; i < 8; ++i) { a[i] = lo[i]; a[i + 8] = hi[i]; }
  return a;
}

// B-matrix 32x16 f16 from N-major (transposed) storage: lane L: col = L%16,
// halves K = (L/16)*8 + {0..7} and +16 (same K order as the A-frag).
DEV v16h load_b_frag(const half_t* base, int ld) {
  int l = threadIdx.x & 31;
  const half_t* p = base + (l & 15) * ld + (l >> 4) * 8;
  v8h lo = *(const v8h*)p;
  v8h hi = *(const v8h*)(p + 16);
  v16h b;
#pragma unroll
  for (int i = 0; i < 8; ++i) { b[i] = lo[i]; b[i + 8] = hi[i]; }
  return b;
}

DEV v8f wmma_f16(v16h a, v16h b, v8f c) {
  return __builtin_amdgcn_wmma_f32_16x16x32_f16(
      /*neg_a=*/false, a, /*neg_b=*/false, b,
      /*c_mod=*/(short)0, c, /*reuse_a=*/false, /*reuse_b=*/false);
}

// Fast activations: single v_exp_f32 each (TRANS op, co-executes with WMMA).
DEV float fast_sigmoid(float x) {
  return __fdividef(1.0f, 1.0f + __expf(-x));
}
DEV float fast_tanh(float x) {
  x = fminf(15.0f, fmaxf(-15.0f, x));
  float e = __expf(2.0f * x);
  return __fdividef(e - 1.0f, e + 1.0f);
}

DEV float wave_reduce_add(float v) {
#pragma unroll
  for (int o = 16; o > 0; o >>= 1) v += __shfl_xor(v, o, 32);
  return v;
}
DEV float wave_reduce_max(float v) {
#pragma unroll
  for (int o = 16; o > 0; o >>= 1) v = fmaxf(v, __shfl_xor(v, o, 32));
  return v;
}

// ---- prep kernels ----------------------------------------------------------
__global__ void k_cast_f16(const float* __restrict__ src, half_t* __restrict__ dst, int n) {
  for (int i = blockIdx.x * blockDim.x + threadIdx.x; i < n; i += gridDim.x * blockDim.x)
    dst[i] = (half_t)src[i];
}

// dst[c*R + r] = src[r*C + c]   (f32 -> f16, N-major weight layout for B-frags)
__global__ void k_transpose_f16(const float* __restrict__ src, half_t* __restrict__ dst,
                                int R, int C) {
  int n = R * C;
  for (int i = blockIdx.x * blockDim.x + threadIdx.x; i < n; i += gridDim.x * blockDim.x) {
    int r = i / C, c = i - r * C;
    dst[c * R + r] = (half_t)src[i];
  }
}

__global__ void k_zero_f16(half_t* __restrict__ dst, int n) {
  for (int i = blockIdx.x * blockDim.x + threadIdx.x; i < n; i += gridDim.x * blockDim.x)
    dst[i] = (half_t)0.f;
}

// ---- xg = x @ Wx + b, stored [T][B][4H] so the LSTM C-loads coalesce -------
__global__ void k_xg_gemm(const half_t* __restrict__ xh, const half_t* __restrict__ WxT,
                          const float* __restrict__ bias, float* __restrict__ xg) {
  int gw = (blockIdx.x * blockDim.x + threadIdx.x) >> 5;  // 16384 waves
  int l  = threadIdx.x & 31;
  int mt = gw >> 5;   // 512 tiles of 16 rows (rows = b*T + t)
  int ng = gw & 31;   // 32 groups of 64 output cols
  v8f acc[4];
#pragma unroll
  for (int s = 0; s < 4; ++s) {
    float bv = bias[ng * 64 + s * 16 + (l & 15)];
#pragma unroll
    for (int r = 0; r < 8; ++r) acc[s][r] = bv;
  }
  const half_t* abase = xh + (size_t)mt * 16 * Dz;
  const half_t* bbase[4];
#pragma unroll
  for (int s = 0; s < 4; ++s) bbase[s] = WxT + (size_t)(ng * 64 + s * 16) * Dz;

  // pipelined: fragments for kt+1 load while kt's WMMAs execute
  v16h a_cur = load_a_frag(abase, Dz);
  v16h b_cur[4];
#pragma unroll
  for (int s = 0; s < 4; ++s) b_cur[s] = load_b_frag(bbase[s], Dz);
#pragma unroll
  for (int kt = 0; kt < 16; ++kt) {
    v16h a_nxt;
    v16h b_nxt[4];
    if (kt < 15) {
      a_nxt = load_a_frag(abase + (kt + 1) * 32, Dz);
#pragma unroll
      for (int s = 0; s < 4; ++s) b_nxt[s] = load_b_frag(bbase[s] + (kt + 1) * 32, Dz);
    }
    pipeline_fence();  // keep kt+1 loads above kt's WMMAs
#pragma unroll
    for (int s = 0; s < 4; ++s) acc[s] = wmma_f16(a_cur, b_cur[s], acc[s]);
    pipeline_fence();
    if (kt < 15) {
      a_cur = a_nxt;
#pragma unroll
      for (int s = 0; s < 4; ++s) b_cur[s] = b_nxt[s];
    }
  }

  int hi = l >> 4, n0 = l & 15;
#pragma unroll
  for (int s = 0; s < 4; ++s) {
    int col = ng * 64 + s * 16 + n0;
#pragma unroll
    for (int r = 0; r < 8; ++r) {
      int in_row = mt * 16 + r + 8 * hi;        // = b*T + t
      int b_ = in_row >> 9, t = in_row & 511;
      xg[(size_t)(t * Bz + b_) * FH + col] = acc[s][r];
    }
  }
}

// ---- LSTM scan: one 32-wave workgroup; wave w owns h columns [16w,16w+16) --
// Software-pipelined: Wh B-fragments are prefetched one k-tile ahead in a
// register ring (t-invariant addresses, so the ring stays primed across the
// per-step barrier), and the next step's xg pre-activations are loaded during
// the current step's gate elementwise phase.
__global__ void __launch_bounds__(1024) k_lstm(const float* __restrict__ xg,
                                               const half_t* __restrict__ WhT,
                                               float* __restrict__ hs) {
  __shared__ __align__(16) half_t hb[2][Bz * Dz];  // 32KB double-buffered h
  int tid = threadIdx.x;
  int w = tid >> 5, l = tid & 31;
  for (int i = tid; i < Bz * Dz; i += 1024) hb[0][i] = (half_t)0.f;
  __syncthreads();

  int hi = l >> 4, n0 = l & 15;
  int col0 = w * 16;

  const half_t* wbase[4];
#pragma unroll
  for (int g = 0; g < 4; ++g) wbase[g] = WhT + (size_t)(g * Dz + col0) * Dz;

  v8f c;
#pragma unroll
  for (int r = 0; r < 8; ++r) c[r] = 0.f;

  // prime: xg pre-activations for t=0, B fragments for kt=0
  v8f xacc[4];
#pragma unroll
  for (int g = 0; g < 4; ++g)
#pragma unroll
    for (int r = 0; r < 8; ++r)
      xacc[g][r] = xg[(size_t)(r + 8 * hi) * FH + g * Dz + col0 + n0];

  v16h bcur[4];
#pragma unroll
  for (int g = 0; g < 4; ++g) bcur[g] = load_b_frag(wbase[g], Dz);

  for (int t = 0; t < Tz; ++t) {
    const half_t* hcur = hb[t & 1];
    half_t* hnxt = hb[(t + 1) & 1];

    v8f acc[4];
#pragma unroll
    for (int g = 0; g < 4; ++g) acc[g] = xacc[g];

    // h_{t-1} @ Wh : 16 k-tiles x 4 gates = 64 WMMAs/wave/step,
    // B fragments double-buffered one k-tile ahead.
#pragma unroll
    for (int kt = 0; kt < 16; ++kt) {
      v16h a = load_a_frag(hcur + kt * 32, Dz);
      int kn = (kt + 1) & 15;  // kt==15 prefetches next step's kt=0 frags
      v16h bnxt[4];
#pragma unroll
      for (int g = 0; g < 4; ++g) bnxt[g] = load_b_frag(wbase[g] + kn * 32, Dz);
      pipeline_fence();  // keep kt+1 loads above kt's WMMAs
#pragma unroll
      for (int g = 0; g < 4; ++g) acc[g] = wmma_f16(a, bcur[g], acc[g]);
      pipeline_fence();
#pragma unroll
      for (int g = 0; g < 4; ++g) bcur[g] = bnxt[g];
    }

    // prefetch next step's xg pre-activations (overlaps elementwise+barrier)
    if (t + 1 < Tz) {
#pragma unroll
      for (int g = 0; g < 4; ++g)
#pragma unroll
        for (int r = 0; r < 8; ++r)
          xacc[g][r] = xg[(size_t)((t + 1) * Bz + r + 8 * hi) * FH + g * Dz + col0 + n0];
    }

    // gates: i,f,g,o
#pragma unroll
    for (int r = 0; r < 8; ++r) {
      float cv = fast_sigmoid(acc[1][r]) * c[r] + fast_sigmoid(acc[0][r]) * fast_tanh(acc[2][r]);
      c[r] = cv;
      float hv = fast_sigmoid(acc[3][r]) * fast_tanh(cv);
      int m = r + 8 * hi;  // batch index
      hnxt[m * Dz + col0 + n0] = (half_t)hv;
      hs[(size_t)(m * Tz + t) * Dz + col0 + n0] = hv;  // [B][T][D]
    }
    __syncthreads();
  }
}

// ---- row LayerNorm (one wave per 512-wide row), optional residual/f16 out --
__global__ void k_ln(const float* __restrict__ in, const float* __restrict__ res,
                     const float* __restrict__ scale, const float* __restrict__ bias,
                     float* __restrict__ outf, half_t* __restrict__ outh, int rows) {
  int w = threadIdx.x >> 5, l = threadIdx.x & 31;
  int row = blockIdx.x * (blockDim.x >> 5) + w;
  if (row >= rows) return;
  const float* p = in + (size_t)row * Dz;
  float v[16];
  float s = 0.f;
#pragma unroll
  for (int i = 0; i < 16; ++i) { v[i] = p[l + 32 * i]; s += v[i]; }
  float mu = wave_reduce_add(s) * (1.0f / Dz);
  float q = 0.f;
#pragma unroll
  for (int i = 0; i < 16; ++i) { float d = v[i] - mu; q += d * d; }
  float rs = rsqrtf(wave_reduce_add(q) * (1.0f / Dz) + 1e-6f);
#pragma unroll
  for (int i = 0; i < 16; ++i) {
    int col = l + 32 * i;
    float y = (v[i] - mu) * rs * scale[col] + bias[col];
    if (res) y += res[(size_t)row * Dz + col];
    outf[(size_t)row * Dz + col] = y;
    if (outh) outh[(size_t)row * Dz + col] = (half_t)y;
  }
}

// ---- q/k/v projections. q,k stored f16 zero-padded [B][H][T][32] (K pad for
// 16x16x32 WMMA scores, q pre-scaled by 1/sqrt(HD)); v stored f16 [B][H][D][T].
__global__ void k_qkv(const half_t* __restrict__ h2h,
                      const half_t* __restrict__ WqT, const half_t* __restrict__ WkT,
                      const half_t* __restrict__ WvT,
                      const float* __restrict__ bq, const float* __restrict__ bk,
                      const float* __restrict__ bv,
                      half_t* __restrict__ qh, half_t* __restrict__ kh,
                      half_t* __restrict__ vT) {
  int gw = (blockIdx.x * blockDim.x + threadIdx.x) >> 5;  // 6144 waves
  int l = threadIdx.x & 31;
  int mt = gw / 12;
  int rem = gw - mt * 12;
  int which = rem >> 2, nt = rem & 3;
  const half_t* Wt = (which == 0) ? WqT : (which == 1) ? WkT : WvT;
  const float*  bb = (which == 0) ? bq  : (which == 1) ? bk  : bv;
  v8f acc;
  float b0 = bb[nt * 16 + (l & 15)];
#pragma unroll
  for (int r = 0; r < 8; ++r) acc[r] = b0;

  const half_t* abase = h2h + (size_t)mt * 16 * Dz;
  const half_t* bbase = Wt + (size_t)(nt * 16) * Dz;
  v16h a_cur = load_a_frag(abase, Dz);
  v16h b_cur = load_b_frag(bbase, Dz);
#pragma unroll
  for (int kt = 0; kt < 16; ++kt) {
    v16h a_nxt, b_nxt;
    if (kt < 15) {
      a_nxt = load_a_frag(abase + (kt + 1) * 32, Dz);
      b_nxt = load_b_frag(bbase + (kt + 1) * 32, Dz);
    }
    pipeline_fence();
    acc = wmma_f16(a_cur, b_cur, acc);
    pipeline_fence();
    if (kt < 15) { a_cur = a_nxt; b_cur = b_nxt; }
  }

  int hi = l >> 4, n0 = l & 15;
  int col = nt * 16 + n0;
  int hd = col >> 3, d = col & 7;
  const float qscale = 0.35355339059327373f;  // 1/sqrt(8)
#pragma unroll
  for (int r = 0; r < 8; ++r) {
    int row = mt * 16 + r + 8 * hi;
    int b_ = row >> 9, t = row & 511;
    float vv = acc[r];
    if (which == 0)      qh[((size_t)(b_ * NHEAD + hd) * Tz + t) * 32 + d] = (half_t)(vv * qscale);
    else if (which == 1) kh[((size_t)(b_ * NHEAD + hd) * Tz + t) * 32 + d] = (half_t)vv;
    else                 vT[((size_t)(b_ * NHEAD + hd) * HD + d) * Tz + t] = (half_t)vv;
  }
}

// ---- attention: one wave per (b,h,16 q-rows); 32KB LDS score strip/wave ----
__global__ void __launch_bounds__(64) k_attn(const half_t* __restrict__ qh,
                                             const half_t* __restrict__ kh,
                                             const half_t* __restrict__ vT,
                                             half_t* __restrict__ ctxh) {
  __shared__ __align__(16) float ssc[2 * 16 * 512];  // 64KB (2 waves/block)
  int w = threadIdx.x >> 5, l = threadIdx.x & 31;
  int gid = blockIdx.x * 2 + w;  // 4096 = B*H*(T/16)
  int qt = gid & 31;
  int bh = gid >> 5;             // b*8 + h
  float* S = ssc + w * (16 * 512);
  int hi = l >> 4, n0 = l & 15;

  // scores (q pre-scaled): 32 WMMAs over k-tiles of 16, B pipelined 1 ahead
  v16h afq = load_a_frag(qh + ((size_t)bh * Tz + qt * 16) * 32, 32);
  const half_t* kbase = kh + (size_t)bh * Tz * 32;
  v16h bf_cur = load_b_frag(kbase, 32);
#pragma unroll 4
  for (int kt = 0; kt < 32; ++kt) {
    v16h bf_nxt;
    if (kt < 31) bf_nxt = load_b_frag(kbase + (size_t)(kt + 1) * 16 * 32, 32);
    pipeline_fence();
    v8f acc;
#pragma unroll
    for (int r = 0; r < 8; ++r) acc[r] = 0.f;
    acc = wmma_f16(afq, bf_cur, acc);
#pragma unroll
    for (int r = 0; r < 8; ++r)
      S[(r + 8 * hi) * 512 + kt * 16 + n0] = acc[r];
    pipeline_fence();
    if (kt < 31) bf_cur = bf_nxt;
  }

  // softmax per row; write f16 attn into the lower half of the same LDS strip
  // (f32 row `row` is fully in registers before f16 row `row` overwrites
  // already-consumed f32 rows < row)
  half_t* Ah = (half_t*)S;
  for (int row = 0; row < 16; ++row) {
    float v[16];
    float mx = -1e30f;
#pragma unroll
    for (int i = 0; i < 16; ++i) { v[i] = S[row * 512 + l + 32 * i]; mx = fmaxf(mx, v[i]); }
    mx = wave_reduce_max(mx);
    float sum = 0.f;
#pragma unroll
    for (int i = 0; i < 16; ++i) { v[i] = __expf(v[i] - mx); sum += v[i]; }
    float rsum = __fdividef(1.0f, wave_reduce_add(sum));
#pragma unroll
    for (int i = 0; i < 16; ++i) Ah[row * 512 + l + 32 * i] = (half_t)(v[i] * rsum);
  }

  // ctx = attn @ v : 16 WMMAs; v B-fragments pipelined 1 ahead; A from LDS.
  // B cols n>=8 read junk but those D cols are never stored.
  const half_t* vbase = vT + (size_t)(bh * HD) * Tz;
  v8f acc;
#pragma unroll
  for (int r = 0; r < 8; ++r) acc[r] = 0.f;
  v16h bv_cur = load_b_frag(vbase, Tz);
#pragma unroll
  for (int kt = 0; kt < 16; ++kt) {
    v16h bv_nxt;
    if (kt < 15) bv_nxt = load_b_frag(vbase + (kt + 1) * 32, Tz);
    v16h a = load_a_frag(Ah + kt * 32, 512);
    pipeline_fence();
    acc = wmma_f16(a, bv_cur, acc);
    pipeline_fence();
    if (kt < 15) bv_cur = bv_nxt;
  }
  if (n0 < HD) {
    int b_ = bh >> 3, h_ = bh & 7;
#pragma unroll
    for (int r = 0; r < 8; ++r) {
      int t = qt * 16 + r + 8 * hi;
      ctxh[((size_t)(b_ * Tz + t)) * QKVz + h_ * HD + n0] = (half_t)acc[r];
    }
  }
}

// ---- pre = sigmoid(h2@Wg + bg) * (ctx@Wo + bo) + h2 ------------------------
__global__ void k_out(const half_t* __restrict__ ctxh, const half_t* __restrict__ WoT,
                      const float* __restrict__ bo,
                      const half_t* __restrict__ h2h, const half_t* __restrict__ WgT,
                      const float* __restrict__ bg, const float* __restrict__ h2,
                      float* __restrict__ pre) {
  int gw = (blockIdx.x * blockDim.x + threadIdx.x) >> 5;  // 16384 waves
  int l = threadIdx.x & 31;
  int mt = gw >> 5, nt = gw & 31;
  v8f ao, gg;
  float bov = bo[nt * 16 + (l & 15)], bgv = bg[nt * 16 + (l & 15)];
#pragma unroll
  for (int r = 0; r < 8; ++r) { ao[r] = bov; gg[r] = bgv; }
#pragma unroll
  for (int kt = 0; kt < 2; ++kt) {   // K = 64 (attn output projection)
    v16h a = load_a_frag(ctxh + (size_t)mt * 16 * QKVz + kt * 32, QKVz);
    v16h b = load_b_frag(WoT + (size_t)(nt * 16) * QKVz + kt * 32, QKVz);
    ao = wmma_f16(a, b, ao);
  }
  // gate projection, K = 512, pipelined 1 k-tile ahead
  const half_t* abase = h2h + (size_t)mt * 16 * Dz;
  const half_t* bbase = WgT + (size_t)(nt * 16) * Dz;
  v16h a_cur = load_a_frag(abase, Dz);
  v16h b_cur = load_b_frag(bbase, Dz);
#pragma unroll
  for (int kt = 0; kt < 16; ++kt) {
    v16h a_nxt, b_nxt;
    if (kt < 15) {
      a_nxt = load_a_frag(abase + (kt + 1) * 32, Dz);
      b_nxt = load_b_frag(bbase + (kt + 1) * 32, Dz);
    }
    pipeline_fence();
    gg = wmma_f16(a_cur, b_cur, gg);
    pipeline_fence();
    if (kt < 15) { a_cur = a_nxt; b_cur = b_nxt; }
  }
  int hi = l >> 4, n0 = l & 15;
#pragma unroll
  for (int r = 0; r < 8; ++r) {
    int row = mt * 16 + r + 8 * hi;
    int col = nt * 16 + n0;
    pre[(size_t)row * Dz + col] = fast_sigmoid(gg[r]) * ao[r] + h2[(size_t)row * Dz + col];
  }
}

// ---------------------------------------------------------------------------
extern "C" void kernel_launch(void* const* d_in, const int* in_sizes, int n_in,
                              void* d_out, int out_size, void* d_ws, size_t ws_size,
                              hipStream_t stream) {
  (void)in_sizes; (void)n_in; (void)out_size; (void)ws_size;
  const float* x    = (const float*)d_in[0];
  const float* Wx   = (const float*)d_in[1];
  const float* Wh   = (const float*)d_in[2];
  const float* b    = (const float*)d_in[3];
  const float* ln1s = (const float*)d_in[4];
  const float* ln1b = (const float*)d_in[5];
  const float* Wq   = (const float*)d_in[6];
  const float* bq   = (const float*)d_in[7];
  const float* Wk   = (const float*)d_in[8];
  const float* bk   = (const float*)d_in[9];
  const float* Wv   = (const float*)d_in[10];
  const float* bv   = (const float*)d_in[11];
  const float* Wo   = (const float*)d_in[12];
  const float* bo   = (const float*)d_in[13];
  const float* Wg   = (const float*)d_in[14];
  const float* bg   = (const float*)d_in[15];
  const float* ln2s = (const float*)d_in[16];
  const float* ln2b = (const float*)d_in[17];

  char* ws = (char*)d_ws;
  size_t off = 0;
  auto alloc = [&](size_t bytes) { size_t o = off; off += (bytes + 255) & ~(size_t)255; return o; };

  half_t* xh   = (half_t*)(ws + alloc((size_t)8192 * 512 * 2));
  half_t* WxT  = (half_t*)(ws + alloc((size_t)2048 * 512 * 2));
  half_t* WhT  = (half_t*)(ws + alloc((size_t)2048 * 512 * 2));
  half_t* WgT  = (half_t*)(ws + alloc((size_t)512 * 512 * 2));
  half_t* WqT  = (half_t*)(ws + alloc((size_t)64 * 512 * 2));
  half_t* WkT  = (half_t*)(ws + alloc((size_t)64 * 512 * 2));
  half_t* WvT  = (half_t*)(ws + alloc((size_t)64 * 512 * 2));
  half_t* WoT  = (half_t*)(ws + alloc((size_t)512 * 64 * 2));
  float*  xg   = (float*)(ws + alloc((size_t)512 * 16 * 2048 * 4));   // [T][B][4H]
  float*  hs   = (float*)(ws + alloc((size_t)8192 * 512 * 4));        // [B][T][D]
  float*  h2   = (float*)(ws + alloc((size_t)8192 * 512 * 4));
  half_t* h2h  = (half_t*)(ws + alloc((size_t)8192 * 512 * 2));
  half_t* qh   = (half_t*)(ws + alloc((size_t)16 * 8 * 512 * 32 * 2)); // [B][H][T][32]
  half_t* kh   = (half_t*)(ws + alloc((size_t)16 * 8 * 512 * 32 * 2));
  half_t* vTt  = (half_t*)(ws + alloc((size_t)16 * 8 * 8 * 512 * 2)); // [B][H][D][T]
  half_t* ctxh = (half_t*)(ws + alloc((size_t)8192 * 64 * 2));        // [B*T][64]
  float*  pre  = (float*)(ws + alloc((size_t)8192 * 512 * 4));

  // 1) pack/cast inputs and weights
  k_cast_f16<<<4096, 256, 0, stream>>>(x, xh, 8192 * 512);
  k_transpose_f16<<<2048, 256, 0, stream>>>(Wx, WxT, 512, 2048);
  k_transpose_f16<<<2048, 256, 0, stream>>>(Wh, WhT, 512, 2048);
  k_transpose_f16<<<512, 256, 0, stream>>>(Wg, WgT, 512, 512);
  k_transpose_f16<<<64, 256, 0, stream>>>(Wq, WqT, 512, 64);
  k_transpose_f16<<<64, 256, 0, stream>>>(Wk, WkT, 512, 64);
  k_transpose_f16<<<64, 256, 0, stream>>>(Wv, WvT, 512, 64);
  k_transpose_f16<<<64, 256, 0, stream>>>(Wo, WoT, 64, 512);
  k_zero_f16<<<2048, 256, 0, stream>>>(qh, 16 * 8 * 512 * 32);
  k_zero_f16<<<2048, 256, 0, stream>>>(kh, 16 * 8 * 512 * 32);

  // 2) xg = x @ Wx + b   (parallel GEMM, WMMA, pipelined)
  k_xg_gemm<<<2048, 256, 0, stream>>>(xh, WxT, b, xg);

  // 3) LSTM scan (single 32-wave workgroup, 2048 WMMAs/step, pipelined)
  k_lstm<<<1, 1024, 0, stream>>>(xg, WhT, hs);

  // 4) h2 = LN1(h) + x  (f32 + f16 copies)
  k_ln<<<1024, 256, 0, stream>>>(hs, x, ln1s, ln1b, h2, h2h, 8192);

  // 5) q/k/v projections
  k_qkv<<<768, 256, 0, stream>>>(h2h, WqT, WkT, WvT, bq, bk, bv, qh, kh, vTt);

  // 6) attention (scores + softmax + ctx)
  k_attn<<<2048, 64, 0, stream>>>(qh, kh, vTt, ctxh);

  // 7) pre = sigmoid(h2@Wg+bg) * (ctx@Wo+bo) + h2
  k_out<<<2048, 256, 0, stream>>>(ctxh, WoT, bo, h2h, WgT, bg, h2, pre);

  // 8) out = LN2(pre)
  k_ln<<<1024, 256, 0, stream>>>(pre, nullptr, ln2s, ln2b, (float*)d_out, nullptr, 8192);
}
